// cata_mamba_80341658239119
// MI455X (gfx1250) — matrix operand order repaired and checked by reading, not compiled, not run on hardware
//
#include <hip/hip_runtime.h>
#include <hip/hip_bf16.h>
#include <math.h>

// ---------------- problem constants (match reference) ----------------
#define D_MODEL   512
#define D_STATE   16
#define D_CONV    4
#define D_INNER   1024          // EXPAND * D_MODEL
#define DT_RANK   32            // (512+15)//16
#define N_LAYERS  2
#define NUM_CLASSES 10
#define BB        4
#define LL        1024
#define M_ROWS    (BB * LL)     // 4096 token rows
#define DBC_COLS  (DT_RANK + 2 * D_STATE)   // 64

typedef __attribute__((ext_vector_type(16))) _Float16 v16h;
typedef __attribute__((ext_vector_type(8)))  _Float16 v8h;
typedef __attribute__((ext_vector_type(4)))  _Float16 v4h;
typedef __attribute__((ext_vector_type(8)))  float    v8f;

#define WMMA_F16(Afrag, Bfrag, Cacc) \
    __builtin_amdgcn_wmma_f32_16x16x32_f16(false, (Afrag), false, (Bfrag), (short)0, (Cacc), false, false)

__device__ __forceinline__ float softplus_f(float v) {
    return fmaxf(v, 0.0f) + __logf(1.0f + __expf(-fabsf(v)));   // branchless
}
__device__ __forceinline__ float silu_f(float v) {
    return v / (1.0f + __expf(-v));
}

// ---------------------------------------------------------------------
// fp32 -> fp16 bulk convert (4 elements / thread)
// ---------------------------------------------------------------------
__global__ void mamba_cvt_f16(_Float16* __restrict__ o, const float* __restrict__ in, int n4)
{
    const int i = blockIdx.x * blockDim.x + threadIdx.x;
    if (i >= n4) return;
    const float4 f = reinterpret_cast<const float4*>(in)[i];
    v4h h;
    h[0] = (_Float16)f.x; h[1] = (_Float16)f.y;
    h[2] = (_Float16)f.z; h[3] = (_Float16)f.w;
    reinterpret_cast<v4h*>(o)[i] = h;
}

// ---------------------------------------------------------------------
// WMMA GEMM on pre-converted fp16 operands.
//   C[M,N]  = act( A[M,lda]h x W[N,K]h^T + bias[N] ) + res[M,N]
//   Ch[M,N] = (fp16)C                                  (if MIRROR)
// One wave owns a 32(M) x 64(N) strip: 2 A-fragments x 4 B-fragments ->
// 8 v_wmma per K-step of 32, software-pipelined (next-step fragments are
// loaded before the current step's wmmas, so loads overlap matrix math).
// Block = up to 4 waves along N (32 x 256 tile).  ACT: 0 none, 1 softplus.
// ---------------------------------------------------------------------
template<int ACT, bool HASB, bool HASR, bool MIRROR>
__global__ __launch_bounds__(128)
void mamba_wmma_gemm(float* __restrict__ C, _Float16* __restrict__ Ch,
                     const _Float16* __restrict__ A, const _Float16* __restrict__ W,
                     const float* __restrict__ bias, const float* __restrict__ res,
                     int M, int N, int K, int lda)
{
    const int lane = threadIdx.x & 31;
    const int wave = threadIdx.x >> 5;
    const int m0 = blockIdx.x * 32;
    const int n0 = blockIdx.y * 256 + wave * 64;
    if (n0 >= N) return;

    // A fragment (16x32 f16): lanes 0-15 -> row, K k..k+15;
    // lanes 16-31 -> same rows, K k+16..k+31.
    const int hiA = (lane & 16) ? 16 : 0;
    // B fragment (32x16 f16): lanes 0-15 -> col, K k..k+15; lanes 16-31 -> K+16..31.
    const int hiB = (lane & 16) ? 16 : 0;

    const _Float16* aRow0 = A + (size_t)(m0 + (lane & 15)) * lda + hiA;
    const _Float16* aRow1 = aRow0 + (size_t)16 * lda;
    const _Float16* wRow  = W + (size_t)(n0 + (lane & 15)) * K + hiB;

    v8f acc[2][4];
#pragma unroll
    for (int i = 0; i < 2; ++i)
#pragma unroll
        for (int j = 0; j < 4; ++j) acc[i][j] = (v8f){};

    union H { v16h v; v8h h[2]; };
    H a0, a1; v16h b0, b1, b2, b3;
    // ---- prologue: fragments for kk = 0 ----
    a0.h[0] = *reinterpret_cast<const v8h*>(aRow0);
    a0.h[1] = *reinterpret_cast<const v8h*>(aRow0 + 8);
    a1.h[0] = *reinterpret_cast<const v8h*>(aRow1);
    a1.h[1] = *reinterpret_cast<const v8h*>(aRow1 + 8);
    b0 = *reinterpret_cast<const v16h*>(wRow);
    b1 = *reinterpret_cast<const v16h*>(wRow + (size_t)16 * K);
    b2 = *reinterpret_cast<const v16h*>(wRow + (size_t)32 * K);
    b3 = *reinterpret_cast<const v16h*>(wRow + (size_t)48 * K);

    for (int kk = 32; kk < K; kk += 32) {
        // issue next-step fragment loads first (overlap with wmmas below)
        H na0, na1; v16h nb0, nb1, nb2, nb3;
        na0.h[0] = *reinterpret_cast<const v8h*>(aRow0 + kk);
        na0.h[1] = *reinterpret_cast<const v8h*>(aRow0 + kk + 8);
        na1.h[0] = *reinterpret_cast<const v8h*>(aRow1 + kk);
        na1.h[1] = *reinterpret_cast<const v8h*>(aRow1 + kk + 8);
        nb0 = *reinterpret_cast<const v16h*>(wRow + kk);
        nb1 = *reinterpret_cast<const v16h*>(wRow + (size_t)16 * K + kk);
        nb2 = *reinterpret_cast<const v16h*>(wRow + (size_t)32 * K + kk);
        nb3 = *reinterpret_cast<const v16h*>(wRow + (size_t)48 * K + kk);
        __builtin_prefetch(aRow0 + kk + 32, 0, 1);        // global_prefetch_b8
        __builtin_prefetch(wRow  + kk + 32, 0, 1);

        acc[0][0] = WMMA_F16(a0.v, b0, acc[0][0]);
        acc[0][1] = WMMA_F16(a0.v, b1, acc[0][1]);
        acc[0][2] = WMMA_F16(a0.v, b2, acc[0][2]);
        acc[0][3] = WMMA_F16(a0.v, b3, acc[0][3]);
        acc[1][0] = WMMA_F16(a1.v, b0, acc[1][0]);
        acc[1][1] = WMMA_F16(a1.v, b1, acc[1][1]);
        acc[1][2] = WMMA_F16(a1.v, b2, acc[1][2]);
        acc[1][3] = WMMA_F16(a1.v, b3, acc[1][3]);

        a0 = na0; a1 = na1; b0 = nb0; b1 = nb1; b2 = nb2; b3 = nb3;
    }
    // ---- epilogue wmmas for the last K-step ----
    acc[0][0] = WMMA_F16(a0.v, b0, acc[0][0]);
    acc[0][1] = WMMA_F16(a0.v, b1, acc[0][1]);
    acc[0][2] = WMMA_F16(a0.v, b2, acc[0][2]);
    acc[0][3] = WMMA_F16(a0.v, b3, acc[0][3]);
    acc[1][0] = WMMA_F16(a1.v, b0, acc[1][0]);
    acc[1][1] = WMMA_F16(a1.v, b1, acc[1][1]);
    acc[1][2] = WMMA_F16(a1.v, b2, acc[1][2]);
    acc[1][3] = WMMA_F16(a1.v, b3, acc[1][3]);

    // D layout: VGPR r -> lanes 0-15: M=r, N=lane; lanes 16-31: M=8+r, N=lane-16.
    const int colb = lane & 15;
#pragma unroll
    for (int mt = 0; mt < 2; ++mt) {
        const int rbase = m0 + mt * 16 + ((lane & 16) ? 8 : 0);
#pragma unroll
        for (int t = 0; t < 4; ++t) {
            const int col = n0 + t * 16 + colb;
            const float bv = HASB ? bias[col] : 0.0f;
#pragma unroll
            for (int r = 0; r < 8; ++r) {
                float v = acc[mt][t][r] + bv;
                if (ACT == 1) v = softplus_f(v);
                if (HASR) v += res[(size_t)(rbase + r) * N + col];
                C[(size_t)(rbase + r) * N + col] = v;
                if (MIRROR) Ch[(size_t)(rbase + r) * N + col] = (_Float16)v;
            }
        }
    }
}

// ---------------------------------------------------------------------
// Depthwise causal conv (width 4) + bias + SiLU over the xc half of xz.
// ---------------------------------------------------------------------
__global__ void mamba_conv_silu(float* __restrict__ xc, _Float16* __restrict__ xch,
                                const float* __restrict__ xz,
                                const float* __restrict__ cw,
                                const float* __restrict__ cb)
{
    const int idx = blockIdx.x * blockDim.x + threadIdx.x;
    if (idx >= M_ROWS * D_INNER) return;
    const int d  = idx & (D_INNER - 1);
    const int bl = idx >> 10;
    const int l  = bl & (LL - 1);
    const int b  = bl >> 10;
    float acc = cb[d];
#pragma unroll
    for (int k = 0; k < D_CONV; ++k) {
        const int ls = l - (D_CONV - 1) + k;
        if (ls >= 0)
            acc += xz[(size_t)(b * LL + ls) * (2 * D_INNER) + d] * cw[d * D_CONV + k];
    }
    const float s = silu_f(acc);
    xc[idx]  = s;
    xch[idx] = (_Float16)s;
}

// ---------------------------------------------------------------------
// Selective scan. Grid = BB * (D_INNER/256) blocks, 256 threads (8 waves).
// Per-step B_t/C_t (128 B) is staged into LDS with the CDNA5 async engine:
// 8 lanes issue global_load_async_to_lds_b128 (tracked by ASYNCcnt), then
// the block waits with s_wait_asynccnt 0 + barrier and reads sBC via ds_load.
// sBC is the only LDS object, so its block-relative byte address is tid*16.
// NOTE: &sBC[0] is passed as an asm input so the compiler sees the shared
// array escape into the asm (which writes it) — otherwise LDS-global
// optimization would prove "never stored" and fold the ds_loads away.
// ---------------------------------------------------------------------
__global__ __launch_bounds__(256)
void mamba_selective_scan(_Float16* __restrict__ yh,
                          const float* __restrict__ dbc,
                          const float* __restrict__ dtb,
                          const float* __restrict__ xc,
                          const float* __restrict__ xz,
                          const float* __restrict__ A_log,
                          const float* __restrict__ Dp)
{
    __shared__ float sBC[2 * D_STATE];     // [0,16) = B_t, [16,32) = C_t
    const int b = blockIdx.x >> 2;                   // D_INNER/256 == 4
    const int d = ((blockIdx.x & 3) << 8) + threadIdx.x;

    float a[D_STATE], h[D_STATE];
#pragma unroll
    for (int n = 0; n < D_STATE; ++n) {
        a[n] = -__expf(A_log[d * D_STATE + n]);
        h[n] = 0.0f;
    }
    const float dval = Dp[d];
    const unsigned ldsOff = (unsigned)(threadIdx.x * 16);   // 16B per lane

    for (int l = 0; l < LL; ++l) {
        const int row = b * LL + l;
        if (threadIdx.x < 8) {
            const float* gsrc = dbc + (size_t)row * DBC_COLS + DT_RANK + threadIdx.x * 4;
            // async copy 16B/lane into LDS; %2 ties sBC's address into the asm
            asm volatile("global_load_async_to_lds_b128 %0, %1, off"
                         :: "v"(ldsOff), "v"(gsrc), "v"(&sBC[0]) : "memory");
        }
        asm volatile("s_wait_asynccnt 0x0" ::: "memory");
        __syncthreads();

        const float dtv = dtb[(size_t)row * D_INNER + d];
        const float xv  = xc [(size_t)row * D_INNER + d];
        const float dtx = dtv * xv;
        float y = 0.0f;
#pragma unroll
        for (int n = 0; n < D_STATE; ++n) {
            const float dA = __expf(dtv * a[n]);
            h[n] = dA * h[n] + dtx * sBC[n];
            y = fmaf(h[n], sBC[D_STATE + n], y);
        }
        const float zv = xz[(size_t)row * (2 * D_INNER) + D_INNER + d];
        yh[(size_t)row * D_INNER + d] = (_Float16)((y + xv * dval) * silu_f(zv));
        __syncthreads();
    }
}

// ---------------------------------------------------------------------
// Classifier head.
// ---------------------------------------------------------------------
__global__ void mamba_build_cls(float* __restrict__ Weff, float* __restrict__ beff,
                                const float* __restrict__ mix_w,
                                const float* __restrict__ cls_w,
                                const float* __restrict__ cls_b)
{
    const int idx = blockIdx.x * blockDim.x + threadIdx.x;
    const float m  = fmaxf(fmaxf(mix_w[0], mix_w[1]), mix_w[2]);
    const float e0 = __expf(mix_w[0] - m), e1 = __expf(mix_w[1] - m), e2 = __expf(mix_w[2] - m);
    const float inv = 1.0f / (e0 + e1 + e2);
    const float w0 = e0 * inv, w1 = e1 * inv, w2 = e2 * inv;
    if (idx < NUM_CLASSES * D_MODEL) {
        Weff[idx] = w0 * cls_w[idx]
                  + w1 * cls_w[NUM_CLASSES * D_MODEL + idx]
                  + w2 * cls_w[2 * NUM_CLASSES * D_MODEL + idx];
    }
    if (idx < NUM_CLASSES) {
        beff[idx] = w0 * cls_b[idx] + w1 * cls_b[NUM_CLASSES + idx]
                  + w2 * cls_b[2 * NUM_CLASSES + idx];
    }
}

__global__ void mamba_cls_head(float* __restrict__ out,
                               const float* __restrict__ x,
                               const float* __restrict__ Weff,
                               const float* __restrict__ beff)
{
    const int idx = blockIdx.x * blockDim.x + threadIdx.x;
    if (idx >= M_ROWS * NUM_CLASSES) return;
    const int c  = idx % NUM_CLASSES;
    const int bl = idx / NUM_CLASSES;
    const float4* xp = reinterpret_cast<const float4*>(x + (size_t)bl * D_MODEL);
    const float4* wp = reinterpret_cast<const float4*>(Weff + (size_t)c * D_MODEL);
    float acc = 0.0f;
#pragma unroll 4
    for (int k = 0; k < D_MODEL / 4; ++k) {
        const float4 xv = xp[k], wv = wp[k];
        acc = fmaf(xv.x, wv.x, acc);
        acc = fmaf(xv.y, wv.y, acc);
        acc = fmaf(xv.z, wv.z, acc);
        acc = fmaf(xv.w, wv.w, acc);
    }
    out[idx] = acc + beff[c];
}

// ---------------------------------------------------------------------
static inline char* ws_take(char*& p, size_t bytes) {
    char* r = p;
    p += (bytes + 255) & ~(size_t)255;     // 256B-align every carve
    return r;
}

extern "C" void kernel_launch(void* const* d_in, const int* in_sizes, int n_in,
                              void* d_out, int out_size, void* d_ws, size_t ws_size,
                              hipStream_t stream)
{
    const float* x_in  = (const float*)d_in[0];
    const float* in_w  = (const float*)d_in[1];
    const float* cw    = (const float*)d_in[2];
    const float* cb    = (const float*)d_in[3];
    const float* xp_w  = (const float*)d_in[4];
    const float* dtp_w = (const float*)d_in[5];
    const float* dtp_b = (const float*)d_in[6];
    const float* A_log = (const float*)d_in[7];
    const float* Dp    = (const float*)d_in[8];
    const float* out_w = (const float*)d_in[9];
    const float* mix_w = (const float*)d_in[10];
    const float* cls_w = (const float*)d_in[11];
    const float* cls_b = (const float*)d_in[12];
    float* out = (float*)d_out;

    // ---- workspace carve-up ----
    char* p = (char*)d_ws;
    float*    xz    = (float*)   ws_take(p, sizeof(float) * M_ROWS * 2 * D_INNER);
    float*    xc    = (float*)   ws_take(p, sizeof(float) * M_ROWS * D_INNER);
    float*    dbc   = (float*)   ws_take(p, sizeof(float) * M_ROWS * DBC_COLS);
    float*    dtb   = (float*)   ws_take(p, sizeof(float) * M_ROWS * D_INNER);
    float*    xA    = (float*)   ws_take(p, sizeof(float) * M_ROWS * D_MODEL);
    float*    xB    = (float*)   ws_take(p, sizeof(float) * M_ROWS * D_MODEL);
    float*    Weff  = (float*)   ws_take(p, sizeof(float) * NUM_CLASSES * D_MODEL);
    float*    beff  = (float*)   ws_take(p, sizeof(float) * NUM_CLASSES);
    _Float16* h_x   = (_Float16*)ws_take(p, sizeof(_Float16) * M_ROWS * D_MODEL);
    _Float16* h_xc  = (_Float16*)ws_take(p, sizeof(_Float16) * M_ROWS * D_INNER);
    _Float16* h_dbc = (_Float16*)ws_take(p, sizeof(_Float16) * M_ROWS * DBC_COLS);
    _Float16* h_y   = (_Float16*)ws_take(p, sizeof(_Float16) * M_ROWS * D_INNER);
    _Float16* h_inw = (_Float16*)ws_take(p, sizeof(_Float16) * N_LAYERS * 2 * D_INNER * D_MODEL);
    _Float16* h_xpw = (_Float16*)ws_take(p, sizeof(_Float16) * N_LAYERS * DBC_COLS * D_INNER);
    _Float16* h_dtw = (_Float16*)ws_take(p, sizeof(_Float16) * N_LAYERS * D_INNER * DT_RANK);
    _Float16* h_ouw = (_Float16*)ws_take(p, sizeof(_Float16) * N_LAYERS * D_MODEL * D_INNER);

    // ---- one-time fp16 conversion of input + all weights ----
    auto cvt = [&](_Float16* o, const float* i, int n) {
        mamba_cvt_f16<<<(n / 4 + 255) / 256, 256, 0, stream>>>(o, i, n / 4);
    };
    cvt(h_x,   x_in,  M_ROWS * D_MODEL);
    cvt(h_inw, in_w,  N_LAYERS * 2 * D_INNER * D_MODEL);
    cvt(h_xpw, xp_w,  N_LAYERS * DBC_COLS * D_INNER);
    cvt(h_dtw, dtp_w, N_LAYERS * D_INNER * DT_RANK);
    cvt(h_ouw, out_w, N_LAYERS * D_MODEL * D_INNER);

    for (int i = 0; i < N_LAYERS; ++i) {
        const float* xin  = (i == 0) ? x_in : xA;
        float*       xout = (i == 0) ? xA   : xB;

        // xz = x @ in_proj_w^T                                    [4096, 2048]
        mamba_wmma_gemm<0, false, false, false><<<dim3(M_ROWS / 32, (2 * D_INNER) / 256), 128, 0, stream>>>(
            xz, nullptr, h_x, h_inw + (size_t)i * 2 * D_INNER * D_MODEL,
            nullptr, nullptr, M_ROWS, 2 * D_INNER, D_MODEL, D_MODEL);

        // xc = silu(conv(xz[:, :1024]) + cb)  (+ fp16 mirror)
        mamba_conv_silu<<<(M_ROWS * D_INNER) / 256, 256, 0, stream>>>(
            xc, h_xc, xz, cw + (size_t)i * D_INNER * D_CONV, cb + (size_t)i * D_INNER);

        // dbc = xc @ x_proj_w^T   (+ fp16 mirror)                 [4096, 64]
        mamba_wmma_gemm<0, false, false, true><<<dim3(M_ROWS / 32, 1), 32, 0, stream>>>(
            dbc, h_dbc, h_xc, h_xpw + (size_t)i * DBC_COLS * D_INNER,
            nullptr, nullptr, M_ROWS, DBC_COLS, D_INNER, D_INNER);

        // dt = softplus(dbc[:, :32] @ dt_proj_w^T + dt_proj_b)    [4096, 1024]
        mamba_wmma_gemm<1, true, false, false><<<dim3(M_ROWS / 32, D_INNER / 256), 128, 0, stream>>>(
            dtb, nullptr, h_dbc, h_dtw + (size_t)i * D_INNER * DT_RANK,
            dtp_b + (size_t)i * D_INNER, nullptr, M_ROWS, D_INNER, DT_RANK, DBC_COLS);

        // y = scan(...) fused (+xc*D) * silu(z)  -> fp16
        mamba_selective_scan<<<BB * (D_INNER / 256), 256, 0, stream>>>(
            h_y, dbc, dtb, xc, xz,
            A_log + (size_t)i * D_INNER * D_STATE, Dp + (size_t)i * D_INNER);

        // x_out = x_in + y @ out_proj_w^T  (+ fp16 mirror -> next layer input)
        mamba_wmma_gemm<0, false, true, true><<<dim3(M_ROWS / 32, D_MODEL / 256), 128, 0, stream>>>(
            xout, h_x, h_y, h_ouw + (size_t)i * D_MODEL * D_INNER,
            nullptr, xin, M_ROWS, D_MODEL, D_INNER, D_INNER);
    }

    mamba_build_cls<<<(NUM_CLASSES * D_MODEL + 255) / 256, 256, 0, stream>>>(
        Weff, beff, mix_w, cls_w, cls_b);
    mamba_cls_head<<<(M_ROWS * NUM_CLASSES + 255) / 256, 256, 0, stream>>>(
        out, xB, Weff, beff);
}
